// KnowledgeCircuit_69904887709891
// MI455X (gfx1250) — compile-verified, matrix operand order checked by reading, not run on hardware
//
#include <hip/hip_runtime.h>

// ---------------------------------------------------------------------------
// KnowledgeCircuit on gfx1250 WMMA.
//   h   = einsum('bsd,ndr,bsn->bsr')  ==  (row-scaled x) @ stacked F, K=N*D
//   out = einsum('bsr,nrd,bsn->bsd')  ==  (row-scaled h) @ stacked R, K=N*R
// Prep kernels pre-convert weights f32 -> bf16 and pre-transpose to [n][col][k]
// so GEMM B-staging is a raw copy: async global->LDS (ASYNCcnt) when the
// builtin exists, register-staged copy otherwise. A is row-scaled + cvt'd on
// the fly. 8 x v_wmma_f32_16x16x32_bf16 per double-barrier round.
// ---------------------------------------------------------------------------

typedef __bf16 bf16_t;
typedef __attribute__((ext_vector_type(16))) __bf16 v16bf;
typedef __attribute__((ext_vector_type(8)))  __bf16 v8bf;
typedef __attribute__((ext_vector_type(8)))  float  v8f;
typedef __attribute__((ext_vector_type(4)))  int    v4i;

typedef __attribute__((address_space(1))) v4i g_v4i;   // global int4
typedef __attribute__((address_space(3))) v4i l_v4i;   // LDS int4

#define BM 128      // block tile rows (tokens)
#define BN 64       // block tile cols
#define BK 64       // k per staging round = 2 WMMA K-steps
#define APITCH 72   // LDS pitch in bf16 (64+8 pad; 144B: 16B-aligned, conflict-free)
#define BPITCH 72
#define NNEUR 16

#if __has_builtin(__builtin_amdgcn_global_load_async_to_lds_b128)
#define KC_ASYNC 1
#else
#define KC_ASYNC 0
#endif

// 16-element bf16 fragment from two 16B LDS chunks at +0 / +16 elements
// (16-bit A layout: lane half selects K 0..7/16..23 vs 8..15/24..31).
__device__ __forceinline__ v16bf load_frag(const bf16_t* p) {
    v8bf lo = *(const v8bf*)(p);
    v8bf hi = *(const v8bf*)(p + 16);
    v16bf r;
#pragma unroll
    for (int i = 0; i < 8; ++i) { r[i] = lo[i]; r[i + 8] = hi[i]; }
    return r;
}

// ---------------- prep: W[n][K][C] f32  ->  WT[n][C][K] bf16 ----------------
__global__ __launch_bounds__(256)
void kc_prep_bt(const float* __restrict__ W, bf16_t* __restrict__ WT, int K, int C)
{
    __shared__ float tile[32][33];
    const int n  = blockIdx.z;
    const int c0 = blockIdx.x * 32;
    const int k0 = blockIdx.y * 32;
    const int tx = threadIdx.x;    // 0..31
    const int ty = threadIdx.y;    // 0..7

    const float* Wn = W + (size_t)n * K * C;
#pragma unroll
    for (int j = 0; j < 4; ++j) {
        int k = ty + j * 8;
        tile[k][tx] = Wn[(size_t)(k0 + k) * C + c0 + tx];
    }
    __syncthreads();
    bf16_t* Tn = WT + (size_t)n * C * K;
#pragma unroll
    for (int j = 0; j < 4; ++j) {
        int c = ty + j * 8;
        Tn[(size_t)(c0 + c) * K + k0 + tx] = (bf16_t)tile[tx][c];
    }
}

// ---- Out[M,NCOL] = sum_n diag(mixw[:,n]) * A[M,KD] @ WT[n][NCOL][KD]^T ----
template <int KD, int NCOL>
__global__ __launch_bounds__(256)
void kc_stage_gemm(const float* __restrict__ A,
                   const bf16_t* __restrict__ WT,   // [NNEUR][NCOL][KD] bf16
                   const float* __restrict__ mixw,  // [M, NNEUR]
                   float* __restrict__ Out)
{
    constexpr int kPerN = KD / BK;       // power of two (16 or 4)
    constexpr int T     = NNEUR * kPerN;

    __shared__ __align__(16) bf16_t As[BM * APITCH];       // [row][k] scaled bf16
    __shared__ __align__(16) bf16_t Bt[2][BN * BPITCH];    // [col][k] double-buffered
    __shared__ float wsh[BM * NNEUR];

    const int tid  = threadIdx.x;
    const int lane = tid & 31;
    const int wave = tid >> 5;
    const int wm   = wave & 3;     // 4 waves along M
    const int wn   = wave >> 2;    // 2 waves along N

    const int rowBase = blockIdx.y * BM;
    const int colBase = blockIdx.x * BN;

#pragma unroll
    for (int i = 0; i < (BM * NNEUR) / 256; ++i) {
        int idx = i * 256 + tid;
        wsh[idx] = mixw[(size_t)rowBase * NNEUR + idx];
    }
    __syncthreads();

    v8f acc00 = {}, acc01 = {}, acc10 = {}, acc11 = {};

    const int lhalf = lane >> 4;
    const int lm    = lane & 15;
    const int kb    = lhalf * 8;
    const int m0    = wm * 32;
    const int c0    = wn * 32;

    // A staging coords (float4 granularity): 128x64 f32 over 8 iters.
    const int arow = tid >> 4;
    const int ac4  = (tid & 15) * 4;

    // B staging coords: tile row (= output col) / 16B chunk along k.
    const int bRow = wave * 8 + (lane >> 3);  // +0 / +4 over 2 issues -> 64 rows
    const int bK16 = lane & 7;

    float4 aReg[8];
#if !KC_ASYNC
    uint4  bReg[2];
#endif

    auto prefetchA = [&](int k0) {
        const float* An = A + (size_t)rowBase * KD + k0;
#pragma unroll
        for (int it = 0; it < 8; ++it)
            aReg[it] = *(const float4*)(An + (size_t)(arow + it * 16) * KD + ac4);
    };

#if KC_ASYNC
    auto issueB = [&](int n, int k0, int buf) {
        const bf16_t* src = WT + ((size_t)n * NCOL + colBase) * KD + k0;
#pragma unroll
        for (int i = 0; i < 2; ++i) {
            int row = bRow + i * 4;
            const bf16_t* g = src + (size_t)row * KD + bK16 * 8;
            bf16_t*       l = &Bt[buf][row * BPITCH + bK16 * 8];
            __builtin_amdgcn_global_load_async_to_lds_b128(
                (g_v4i*)g, (l_v4i*)l, 0, 0);
        }
    };
#else
    auto prefetchB = [&](int n, int k0) {
        const bf16_t* src = WT + ((size_t)n * NCOL + colBase) * KD + k0;
#pragma unroll
        for (int i = 0; i < 2; ++i)
            bReg[i] = *(const uint4*)(src + (size_t)(bRow + i * 4) * KD + bK16 * 8);
    };
#endif

    int cur_n = 0, cur_k0 = 0;
    float wreg[8];
    prefetchA(0);
#if KC_ASYNC
    issueB(0, 0, 0);
#else
    prefetchB(0, 0);
#endif

    for (int t = 0; t < T; ++t) {
        if ((t & (kPerN - 1)) == 0) {
#pragma unroll
            for (int it = 0; it < 8; ++it)
                wreg[it] = wsh[(arow + it * 16) * NNEUR + cur_n];
        }

        __syncthreads();   // prior round's fragment reads complete

        // ---- Commit A tile: scale rows by mixture weight, cvt bf16 ----
#pragma unroll
        for (int it = 0; it < 8; ++it) {
            bf16_t* dst = &As[(arow + it * 16) * APITCH + ac4];
            float4  v   = aReg[it];
            float   w   = wreg[it];
            dst[0] = (bf16_t)(v.x * w);
            dst[1] = (bf16_t)(v.y * w);
            dst[2] = (bf16_t)(v.z * w);
            dst[3] = (bf16_t)(v.w * w);
        }
#if KC_ASYNC
        // Complete this round's async B tile (issued last iteration).
        asm volatile("s_wait_asynccnt 0x0" ::: "memory");
#else
#pragma unroll
        for (int i = 0; i < 2; ++i)
            *(uint4*)&Bt[t & 1][(bRow + i * 4) * BPITCH + bK16 * 8] = bReg[i];
#endif
        __syncthreads();

        // ---- Kick off next tile's loads; overlap with the WMMAs below ----
        int nn = cur_n, nk = cur_k0 + BK;
        if (nk == KD) { nk = 0; ++nn; }
        if (t + 1 < T) {
            prefetchA(nk);
#if KC_ASYNC
            issueB(nn, nk, (t + 1) & 1);
#else
            prefetchB(nn, nk);
#endif
        }

        const bf16_t* BtC = Bt[t & 1];

        // ---- 2 K-steps x (2x2) WMMA ----
#pragma unroll
        for (int s = 0; s < 2; ++s) {
            const int ko = s * 32 + kb;
            v16bf a0 = load_frag(&As[(m0 + lm)       * APITCH + ko]);
            v16bf a1 = load_frag(&As[(m0 + 16 + lm)  * APITCH + ko]);
            v16bf b0 = load_frag(&BtC[(c0 + lm)      * BPITCH + ko]);
            v16bf b1 = load_frag(&BtC[(c0 + 16 + lm) * BPITCH + ko]);

            acc00 = __builtin_amdgcn_wmma_f32_16x16x32_bf16(false, a0, false, b0, (short)0, acc00, false, false);
            acc01 = __builtin_amdgcn_wmma_f32_16x16x32_bf16(false, a0, false, b1, (short)0, acc01, false, false);
            acc10 = __builtin_amdgcn_wmma_f32_16x16x32_bf16(false, a1, false, b0, (short)0, acc10, false, false);
            acc11 = __builtin_amdgcn_wmma_f32_16x16x32_bf16(false, a1, false, b1, (short)0, acc11, false, false);
        }

        cur_n = nn; cur_k0 = nk;
    }

    // ---- Writeout: f32 C/D layout: VGPR v -> M = v + 8*lhalf, N = lm ----
    const int outRow0 = rowBase + m0 + lhalf * 8;
    const int outCol  = colBase + c0 + lm;
#pragma unroll
    for (int v = 0; v < 8; ++v) {
        Out[(size_t)(outRow0 + v)      * NCOL + outCol]      = acc00[v];
        Out[(size_t)(outRow0 + v)      * NCOL + outCol + 16] = acc01[v];
        Out[(size_t)(outRow0 + 16 + v) * NCOL + outCol]      = acc10[v];
        Out[(size_t)(outRow0 + 16 + v) * NCOL + outCol + 16] = acc11[v];
    }
}

extern "C" void kernel_launch(void* const* d_in, const int* in_sizes, int n_in,
                              void* d_out, int out_size, void* d_ws, size_t ws_size,
                              hipStream_t stream) {
    (void)in_sizes; (void)n_in; (void)out_size; (void)ws_size;

    const float* x  = (const float*)d_in[0];  // [4,2048,1024]
    const float* fk = (const float*)d_in[1];  // [16,1024,256]
    const float* rk = (const float*)d_in[2];  // [16,256,1024]
    const float* fw = (const float*)d_in[3];  // [4,2048,16]
    const float* rw = (const float*)d_in[4];  // [4,2048,16]
    float* out = (float*)d_out;               // [4,2048,1024]

    const int M = 4 * 2048, D = 1024, R = 256;

    // Workspace layout: h (8MB f32) | fkT bf16 (8MB) | rkT bf16 (8MB)
    float*  h   = (float*)d_ws;
    bf16_t* fkT = (bf16_t*)((char*)d_ws + (size_t)M * R * 4);
    bf16_t* rkT = (bf16_t*)((char*)d_ws + (size_t)M * R * 4 + (size_t)NNEUR * D * R * 2);

    dim3 pblk(32, 8);
    // fk: [n][K=D][C=R] -> fkT [n][R][D]
    kc_prep_bt<<<dim3(R / 32, D / 32, NNEUR), pblk, 0, stream>>>(fk, fkT, D, R);
    // rk: [n][K=R][C=D] -> rkT [n][D][R]
    kc_prep_bt<<<dim3(D / 32, R / 32, NNEUR), pblk, 0, stream>>>(rk, rkT, R, D);

    dim3 block(256);
    // Stage 1: h = (scaled x) @ F   -> [M, R]
    kc_stage_gemm<D, R><<<dim3(R / BN, M / BM), block, 0, stream>>>(x, fkT, fw, h);
    // Stage 2: out = (scaled h) @ Rk -> [M, D]
    kc_stage_gemm<R, D><<<dim3(D / BN, M / BM), block, 0, stream>>>(h, rkT, rw, out);
}
